// MMoE_83502754169016
// MI455X (gfx1250) — matrix operand order repaired and checked by reading, not compiled, not run on hardware
//
#include <hip/hip_runtime.h>
#include <hip/hip_bf16.h>
#include <stdint.h>

// ---------------- problem constants ----------------
#define B_TOT   512
#define S_LEN   256
#define C_N     8
#define ENC_IN  2580
#define TILE    16          // batch samples per workgroup
#define NCOL    1280        // 20 mem slots * 64 encode outputs
#define NFRAG   320         // 80 n-tiles * 4 k-steps (memory-encode GEMM)
#define THREADS 256

typedef __attribute__((ext_vector_type(16))) __bf16 v16b;
typedef __attribute__((ext_vector_type(8)))  float  v8f;

union Frag { v16b v; unsigned short u[16]; unsigned int d[8]; uint4 q[2]; };

__device__ __forceinline__ unsigned short f2bf(float f) {
  union { float f; unsigned u; } x; x.f = f;
  unsigned r = x.u + 0x7FFFu + ((x.u >> 16) & 1u);   // round-to-nearest-even
  return (unsigned short)(r >> 16);
}
__device__ __forceinline__ float fast_rcp(float x) { return __builtin_amdgcn_rcpf(x); }
__device__ __forceinline__ float sigm(float x) { return fast_rcp(1.0f + __expf(-x)); }
__device__ __forceinline__ float tanh_fast(float x) {   // branch-free, native exp/rcp
  float xc = fminf(15.f, fmaxf(-15.f, x));
  float t  = __expf(2.0f * xc);
  return (t - 1.0f) * fast_rcp(t + 1.0f);
}

// ---------------- prep: swizzle W1 into WMMA-B fragments ----------------
// Bmat[K=0..127][ncol=0..1279], ncol = m*64+nn, Bmat[K][ncol] = W1[nn][m*129+K].
__global__ void prep_w1r(const float* __restrict__ W1, unsigned short* __restrict__ w1r) {
  int idx  = blockIdx.x * THREADS + threadIdx.x;   // 0..163839
  int f    = idx >> 9;                             // fragment 0..319
  int rem  = idx & 511;
  int lane = rem >> 4;
  int i    = rem & 15;
  int j = f >> 2, kk = f & 3;
  int p = i >> 1, bit = i & 1;
  int klocal = ((p < 4) ? 0 : 16) + 8 * (lane >> 4) + 2 * (p & 3) + bit;     // dense 32x16 16-bit B layout
  int K    = 32 * kk + klocal;
  int ncol = 16 * j + (lane & 15);
  int m = ncol >> 6, nn = ncol & 63;
  w1r[idx] = f2bf(W1[nn * ENC_IN + m * 129 + K]);
}

// expert-recurrent B: Bexp[K][n] = (K<16) ? W_hh[n][K] : 0   (n = c*64+g)
__global__ void prep_wexp(const float* __restrict__ W_hh, unsigned short* __restrict__ wexp) {
  int idx  = blockIdx.x * THREADS + threadIdx.x;   // 0..16383
  int f    = idx >> 9;
  int rem  = idx & 511;
  int lane = rem >> 4;
  int i    = rem & 15;
  int p = i >> 1, bit = i & 1;
  int kl = ((p < 4) ? 0 : 16) + 8 * (lane >> 4) + 2 * (p & 3) + bit;
  int n  = 16 * f + (lane & 15);
  float v = (kl < 16) ? W_hh[n * 16 + kl] : 0.f;
  wexp[idx] = f2bf(v);
}

// agent LSTM B: per layer l, B[k][n] = k<16 ? Wa_ih[l][n][k] : Wa_hh[l][n][k-16]
__global__ void prep_wagt(const float* __restrict__ Wa_ih, const float* __restrict__ Wa_hh,
                          unsigned short* __restrict__ wagt) {
  int idx  = blockIdx.x * THREADS + threadIdx.x;   // 0..4095
  int f    = idx >> 9;                             // 0..7: l = f>>2, tile = f&3
  int rem  = idx & 511;
  int lane = rem >> 4;
  int i    = rem & 15;
  int p = i >> 1, bit = i & 1;
  int kl = ((p < 4) ? 0 : 16) + 8 * (lane >> 4) + 2 * (p & 3) + bit;
  int l = f >> 2, qt = f & 3;
  int n = 16 * qt + (lane & 15);                   // 0..63
  float v = (kl < 16) ? Wa_ih[(l * 64 + n) * 16 + kl]
                      : Wa_hh[(l * 64 + n) * 16 + (kl - 16)];
  wagt[idx] = f2bf(v);
}

// encode-tail B: B[k][n] = W2[n][k], K=0..63 (2 k-steps), N=0..15
__global__ void prep_w2r(const float* __restrict__ W2, unsigned short* __restrict__ w2r) {
  int idx  = blockIdx.x * THREADS + threadIdx.x;   // 0..1023
  int kk   = idx >> 9;
  int rem  = idx & 511;
  int lane = rem >> 4;
  int i    = rem & 15;
  int p = i >> 1, bit = i & 1;
  int kl = ((p < 4) ? 0 : 16) + 8 * (lane >> 4) + 2 * (p & 3) + bit;
  int K  = 32 * kk + kl;
  int n  = lane & 15;
  w2r[idx] = f2bf(W2[n * 64 + K]);
}

// error-channel column of W1 (rank-1 path) + initial-memory ring contributions
__global__ void prep_misc(const float* __restrict__ W1,
                          float* __restrict__ werr, float* __restrict__ initc) {
  for (int ncol = threadIdx.x; ncol < NCOL; ncol += THREADS) {
    int m = ncol >> 6, nn = ncol & 63;
    werr[ncol] = W1[nn * ENC_IN + m * 129 + 128];
  }
  for (int idx = threadIdx.x; idx < 20 * 64; idx += THREADS) {
    int t = idx >> 6, nn = idx & 63;
    float s = 0.f;
    for (int m = t + 1; m < 20; ++m) s += W1[nn * ENC_IN + m * 129 + 128];
    initc[idx] = 0.5f * s;   // init memory rows have 0.5 in the error channel
  }
}

// ---------------- main recurrent kernel ----------------
struct SM {
  float ring[20 * TILE * 64];        // ring accumulators [slot][s][nn]  (80KB)
  float zbuf[TILE * 512];            // pre-activation scratch (reused)  (32KB)
  float nh[TILE * 128];              // expert hidden f32  [s][c*16+h]
  unsigned short nhb[TILE * 128];    // expert hidden bf16 (A-fragment source)
  float nc[TILE * 128];
  float eh[TILE * 16], ec[TILE * 16];
  float xs[TILE], err[TILE];
  float hist[TILE * 20];             // error history ring (for theta)
  float e1[TILE * 64];
  float ai[TILE * 16];
  float ah[2 * TILE * 16], ac[2 * TILE * 16];
  float tAh[TILE * 16], tAc[TILE * 16], tBh[TILE * 16], tBc[TILE * 16];
  float gate[TILE * 8];
  float ocv[TILE * 8], lgt[TILE * 8];
  float pred[TILE];
  // cached weights (small)
  float werr[NCOL];
  float Wih[C_N * 64];
  float bz[C_N * 64];
  float Wo[C_N * 16];
  float bo[C_N];
  float b2[16];
  float b1v[64];
  float Wg[C_N * 16];
  float bg[C_N];
  float ba[2 * 64];
};

__global__ void __launch_bounds__(THREADS, 1)
mmoe_kernel(const float* __restrict__ x, const float* __restrict__ pred0,
            const float* __restrict__ gate0,
            const float* __restrict__ W_ih,
            const float* __restrict__ b_ih, const float* __restrict__ b_hh,
            const float* __restrict__ W_o, const float* __restrict__ b_o,
            const float* __restrict__ b1,  const float* __restrict__ b2,
            const float* __restrict__ Wg,  const float* __restrict__ bg,
            const float* __restrict__ ba_ih, const float* __restrict__ ba_hh,
            const unsigned short* __restrict__ w1r,
            const unsigned short* __restrict__ wexp,
            const unsigned short* __restrict__ wagt,
            const unsigned short* __restrict__ w2r,
            const float* __restrict__ werr_g, const float* __restrict__ initc,
            float* __restrict__ out)
{
  __shared__ SM sm;
  const int tid  = threadIdx.x;
  const int lane = tid & 31, wave = tid >> 5;
  const int b0   = blockIdx.x * TILE;
  const int hi16 = lane >> 4;          // 0/1: which half of the wave
  const int lo16 = lane & 15;
  const int ncbase = 16 * wave + lo16; // per-thread GEMM column base (0..127)
  const int mbase  = ncbase >> 6;      // mem-slot offset of tile q=0
  const int nnk    = ncbase & 63;      // n within slot (q-invariant)

  // -------- load weight caches & init state --------
  for (int i = tid; i < NCOL; i += THREADS) sm.werr[i] = werr_g[i];
  for (int i = tid; i < C_N * 64; i += THREADS) { sm.Wih[i] = W_ih[i]; sm.bz[i] = b_ih[i] + b_hh[i]; }
  for (int i = tid; i < C_N * 16; i += THREADS) { sm.Wo[i] = W_o[i]; sm.Wg[i] = Wg[i]; }
  for (int i = tid; i < C_N; i += THREADS) { sm.bo[i] = b_o[i]; sm.bg[i] = bg[i]; }
  for (int i = tid; i < 16; i += THREADS) sm.b2[i] = b2[i];
  for (int i = tid; i < 64; i += THREADS) sm.b1v[i] = b1[i];
  for (int i = tid; i < 2 * 64; i += THREADS) sm.ba[i] = ba_ih[i] + ba_hh[i];
  for (int i = tid; i < 20 * TILE * 64; i += THREADS) {
    int r = i / (TILE * 64);
    sm.ring[i] = initc[r * 64 + (i & 63)];
  }
  for (int i = tid; i < TILE * 16; i += THREADS) { sm.eh[i] = 0.f; sm.ec[i] = 0.f; }
  for (int i = tid; i < 2 * TILE * 16; i += THREADS) { sm.ah[i] = 0.f; sm.ac[i] = 0.f; }
  for (int i = tid; i < TILE * 20; i += THREADS) sm.hist[i] = 0.5f;
  if (tid < TILE) sm.pred[tid] = pred0[b0 + tid];
  for (int i = tid; i < TILE * 8; i += THREADS)
    sm.gate[i] = gate0[(b0 + (i >> 3)) * 8 + (i & 7)];
  __syncthreads();

  int tmod = 0;                        // t % 20, maintained incrementally
  for (int t = 0; t < S_LEN; ++t) {
    // ---- A: input & error ----
    if (tid < TILE) {
      float xv = x[(b0 + tid) * S_LEN + t];
      sm.xs[tid] = xv;
      float e = xv - sm.pred[tid];
      sm.err[tid] = e;
      sm.hist[tid * 20 + tmod] = e;
    }
    __syncthreads();

    // ---- B0: expert pre-activations via WMMA: z = eh_pad @ Bexp + (bz + x*Wih) ----
    {
      Frag eb[4];
      #pragma unroll
      for (int q2 = 0; q2 < 4; ++q2) {   // issue B loads first (L2 hits)
        const uint4* bp = (const uint4*)(wexp + (((size_t)(wave * 4 + q2)) << 9) + ((size_t)lane << 4));
        eb[q2].q[0] = bp[0]; eb[q2].q[1] = bp[1];
      }
      Frag ea;
      #pragma unroll
      for (int i = 0; i < 16; ++i) {     // ISA 7.12.2 16-bit A layout; K>=16 zero pad
        int p = i >> 1, bit = i & 1;
        int kl = ((p < 4) ? 0 : 16) + 8 * hi16 + 2 * (p & 3) + bit;
        ea.u[i] = (kl < 16) ? f2bf(sm.eh[lo16 * 16 + kl]) : (unsigned short)0;
      }
      #pragma unroll
      for (int q2 = 0; q2 < 4; ++q2) {
        int n = 16 * (wave * 4 + q2) + lo16;
        float biasn = sm.bz[n];
        float win   = sm.Wih[n];
        v8f cini;
        #pragma unroll
        for (int v = 0; v < 8; ++v) cini[v] = biasn + sm.xs[v + 8 * hi16] * win;
        v8f z = __builtin_amdgcn_wmma_f32_16x16x32_bf16(
                  false, ea.v, false, eb[q2].v, (short)0, cini, false, false);
        #pragma unroll
        for (int v = 0; v < 8; ++v) sm.zbuf[(v + 8 * hi16) * 512 + n] = z[v];
      }
    }
    __syncthreads();

    // ---- B1: expert cell nonlinearity (f32), thread = (s,h) ----
    {
      int s = tid >> 4, h = tid & 15;
      float ecv = sm.ec[s * 16 + h];
      #pragma unroll
      for (int c = 0; c < 8; ++c) {
        const float* zr = &sm.zbuf[s * 512 + c * 64];
        float zi = zr[h], zf = zr[16 + h], zgg = zr[32 + h], zo = zr[48 + h];
        float c2 = sigm(zf) * ecv + sigm(zi) * tanh_fast(zgg);
        float hv = sigm(zo) * tanh_fast(c2);
        sm.nh[s * 128 + c * 16 + h]  = hv;
        sm.nhb[s * 128 + c * 16 + h] = f2bf(hv);
        sm.nc[s * 128 + c * 16 + h]  = c2;
      }
    }
    __syncthreads();

    // ---- C: memory-encode GEMM [16x128]@[128x1280], double-buffered bf16 WMMA ----
    {
      Frag afr[4];
      const unsigned int* nhb32 = (const unsigned int*)sm.nhb;
      #pragma unroll
      for (int kk = 0; kk < 4; ++kk) {   // A: bf16 pairs contiguous -> 8 dword loads
        #pragma unroll
        for (int p = 0; p < 8; ++p) {
          int kl = ((p < 4) ? 0 : 16) + 8 * hi16 + 2 * (p & 3);
          afr[kk].d[p] = nhb32[(lo16 * 128 + 32 * kk + kl) >> 1];
        }
      }
      float errv[8];
      #pragma unroll
      for (int v = 0; v < 8; ++v) errv[v] = sm.err[v + 8 * hi16];

      Frag bufA[4], bufB[4];
      auto prefetch = [&](int q, Frag (&buf)[4]) {
        int j = q * 8 + wave;
        #pragma unroll
        for (int kk = 0; kk < 4; ++kk) {
          const uint4* bp = (const uint4*)(w1r + (((size_t)(j * 4 + kk)) << 9) + ((size_t)lane << 4));
          buf[kk].q[0] = bp[0]; buf[kk].q[1] = bp[1];
        }
      };
      auto compute = [&](int q, Frag (&buf)[4]) {
        v8f acc = {0.f, 0.f, 0.f, 0.f, 0.f, 0.f, 0.f, 0.f};
        #pragma unroll
        for (int kk = 0; kk < 4; ++kk)
          acc = __builtin_amdgcn_wmma_f32_16x16x32_bf16(
                  false, afr[kk].v, false, buf[kk].v, (short)0, acc, false, false);
        int rr = tmod + 2 * q + mbase;           // ring slot (t + mslot) % 20
        if (rr >= 20) rr -= 20;
        int base = rr * (TILE * 64) + nnk;
        float we = sm.werr[128 * q + ncbase];
        #pragma unroll
        for (int v = 0; v < 8; ++v)
          sm.ring[base + (v + 8 * hi16) * 64] += acc[v] + errv[v] * we;
      };

      prefetch(0, bufA);
      #pragma unroll 1
      for (int qq = 0; qq < 5; ++qq) {
        prefetch(2 * qq + 1, bufB);
        compute(2 * qq, bufA);
        if (qq < 4) prefetch(2 * qq + 2, bufA);
        compute(2 * qq + 1, bufB);
      }
    }
    __syncthreads();

    // ---- D1: e1 = relu(ring[t%20] + b1), then reset slot for reuse at t+20 ----
    {
      #pragma unroll
      for (int r = 0; r < 4; ++r) {
        int idx = tid + THREADS * r;             // (s,n)
        int s = idx >> 6, n = idx & 63;
        int a = (tmod * TILE + s) * 64 + n;
        float v = sm.ring[a] + sm.b1v[n];
        sm.e1[idx] = v > 0.f ? v : 0.f;
        sm.ring[a] = 0.f;
      }
    }
    __syncthreads();

    // ---- D2: ai = relu(e1 @ W2^T + b2) via WMMA (wave 0) ----
    if (wave == 0) {
      Frag b2f[2];
      #pragma unroll
      for (int kk = 0; kk < 2; ++kk) {
        const uint4* bp = (const uint4*)(w2r + (((size_t)kk) << 9) + ((size_t)lane << 4));
        b2f[kk].q[0] = bp[0]; b2f[kk].q[1] = bp[1];
      }
      v8f acc;
      #pragma unroll
      for (int v = 0; v < 8; ++v) acc[v] = sm.b2[lo16];
      #pragma unroll
      for (int kk = 0; kk < 2; ++kk) {
        Frag afr;
        #pragma unroll
        for (int i = 0; i < 16; ++i) {
          int p = i >> 1, bit = i & 1;
          int kl = ((p < 4) ? 0 : 16) + 8 * hi16 + 2 * (p & 3) + bit;
          afr.u[i] = f2bf(sm.e1[lo16 * 64 + 32 * kk + kl]);
        }
        acc = __builtin_amdgcn_wmma_f32_16x16x32_bf16(
                false, afr.v, false, b2f[kk].v, (short)0, acc, false, false);
      }
      #pragma unroll
      for (int v = 0; v < 8; ++v) {
        float a = acc[v];
        sm.ai[(v + 8 * hi16) * 16 + lo16] = a > 0.f ? a : 0.f;
      }
    }
    __syncthreads();

    // ---- agent LSTM layer 0: z = [ai|ah0] @ [Wa_ih0;Wa_hh0] + ba0 (waves 0-3) ----
    if (wave < 4) {
      Frag bfr;
      const uint4* bp = (const uint4*)(wagt + (((size_t)wave) << 9) + ((size_t)lane << 4));
      bfr.q[0] = bp[0]; bfr.q[1] = bp[1];
      Frag afr;
      #pragma unroll
      for (int i = 0; i < 16; ++i) {
        int p = i >> 1, bit = i & 1;
        int kl = ((p < 4) ? 0 : 16) + 8 * hi16 + 2 * (p & 3) + bit;
        float v = (kl < 16) ? sm.ai[lo16 * 16 + kl] : sm.ah[lo16 * 16 + (kl - 16)];
        afr.u[i] = f2bf(v);
      }
      int n = 16 * wave + lo16;                  // 0..63
      float bn = sm.ba[n];
      v8f cini;
      #pragma unroll
      for (int v = 0; v < 8; ++v) cini[v] = bn;
      v8f z = __builtin_amdgcn_wmma_f32_16x16x32_bf16(
                false, afr.v, false, bfr.v, (short)0, cini, false, false);
      #pragma unroll
      for (int v = 0; v < 8; ++v) sm.zbuf[(v + 8 * hi16) * 64 + n] = z[v];
    }
    __syncthreads();
    {   // layer-0 nonlinearity, thread = (s,h)
      int s = tid >> 4, h = tid & 15;
      const float* zr = &sm.zbuf[s * 64];
      float zi = zr[h], zf = zr[16 + h], zgg = zr[32 + h], zo = zr[48 + h];
      float c2 = sigm(zf) * sm.ac[s * 16 + h] + sigm(zi) * tanh_fast(zgg);
      sm.tAh[s * 16 + h] = sigm(zo) * tanh_fast(c2);
      sm.tAc[s * 16 + h] = c2;
    }
    __syncthreads();

    // ---- agent LSTM layer 1 ----
    if (wave < 4) {
      Frag bfr;
      const uint4* bp = (const uint4*)(wagt + (((size_t)(4 + wave)) << 9) + ((size_t)lane << 4));
      bfr.q[0] = bp[0]; bfr.q[1] = bp[1];
      Frag afr;
      #pragma unroll
      for (int i = 0; i < 16; ++i) {
        int p = i >> 1, bit = i & 1;
        int kl = ((p < 4) ? 0 : 16) + 8 * hi16 + 2 * (p & 3) + bit;
        float v = (kl < 16) ? sm.tAh[lo16 * 16 + kl] : sm.ah[TILE * 16 + lo16 * 16 + (kl - 16)];
        afr.u[i] = f2bf(v);
      }
      int n = 16 * wave + lo16;
      float bn = sm.ba[64 + n];
      v8f cini;
      #pragma unroll
      for (int v = 0; v < 8; ++v) cini[v] = bn;
      v8f z = __builtin_amdgcn_wmma_f32_16x16x32_bf16(
                false, afr.v, false, bfr.v, (short)0, cini, false, false);
      #pragma unroll
      for (int v = 0; v < 8; ++v) sm.zbuf[(v + 8 * hi16) * 64 + n] = z[v];
    }
    __syncthreads();
    {   // layer-1 nonlinearity
      int s = tid >> 4, h = tid & 15;
      const float* zr = &sm.zbuf[s * 64];
      float zi = zr[h], zf = zr[16 + h], zgg = zr[32 + h], zo = zr[48 + h];
      float c2 = sigm(zf) * sm.ac[TILE * 16 + s * 16 + h] + sigm(zi) * tanh_fast(zgg);
      sm.tBh[s * 16 + h] = sigm(zo) * tanh_fast(c2);
      sm.tBc[s * 16 + h] = c2;
    }
    __syncthreads();

    // ---- E0: (s,c) dots for o and gate logits || agent-state commit ----
    if (tid < 128) {
      int s = tid >> 3, c = tid & 7;
      float oc = sm.bo[c], lg = sm.bg[c];
      const float* nhp = &sm.nh[s * 128 + c * 16];
      const float* wo  = &sm.Wo[c * 16];
      const float* hh  = &sm.tBh[s * 16];
      const float* wg  = &sm.Wg[c * 16];
      #pragma unroll
      for (int h2 = 0; h2 < 16; ++h2) { oc += nhp[h2] * wo[h2]; lg += hh[h2] * wg[h2]; }
      sm.ocv[s * 8 + c] = oc;
      sm.lgt[s * 8 + c] = lg;
    } else {
      int u = tid - 128;
      #pragma unroll
      for (int k = 0; k < 2; ++k) {
        int sh = u * 2 + k;
        int s = sh >> 4, h = sh & 15;
        sm.ah[s * 16 + h] = sm.tAh[s * 16 + h];
        sm.ac[s * 16 + h] = sm.tAc[s * 16 + h];
        sm.ah[TILE * 16 + s * 16 + h] = sm.tBh[s * 16 + h];
        sm.ac[TILE * 16 + s * 16 + h] = sm.tBc[s * 16 + h];
      }
    }
    __syncthreads();

    // ---- E1: softmax + theta blend + pred (one thread per sample) ----
    if (tid < TILE) {
      int s = tid;
      float lg[8], mx = -1e30f;
      #pragma unroll
      for (int c = 0; c < 8; ++c) { lg[c] = sm.lgt[s * 8 + c]; mx = fmaxf(mx, lg[c]); }
      float sum = 0.f;
      #pragma unroll
      for (int c = 0; c < 8; ++c) { lg[c] = __expf(lg[c] - mx); sum += lg[c]; }
      float inv = fast_rcp(sum);
      float th = 0.f;
      int tm = tmod;
      #pragma unroll
      for (int m = 0; m < 10; ++m) {
        float e = (t - m >= 0) ? sm.hist[s * 20 + tm] : 0.5f;
        th += fabsf(e);
        tm = (tm == 0) ? 19 : tm - 1;
      }
      th = fminf(1.f, fmaxf(0.f, th * (2.5f / 10.f)));
      float pr = 0.f;
      #pragma unroll
      for (int c = 0; c < 8; ++c) {
        float g = lg[c] * inv * th + sm.gate[s * 8 + c] * (1.f - th);
        sm.gate[s * 8 + c] = g;
        pr += g * sm.ocv[s * 8 + c];
      }
      sm.pred[s] = pr;
      out[(b0 + s) * S_LEN + t] = pr;
    }
    __syncthreads();

    // ---- F: combined expert state eh/ec = gate-weighted mix ----
    {
      int s = tid >> 4, h = tid & 15;
      float he = 0.f, ce = 0.f;
      #pragma unroll
      for (int c = 0; c < 8; ++c) {
        float g = sm.gate[s * 8 + c];
        he += g * sm.nh[s * 128 + c * 16 + h];
        ce += g * sm.nc[s * 128 + c * 16 + h];
      }
      sm.eh[s * 16 + h] = he;
      sm.ec[s * 16 + h] = ce;
    }
    __syncthreads();

    tmod = (tmod == 19) ? 0 : tmod + 1;
  }
}

// ---------------- host launcher ----------------
extern "C" void kernel_launch(void* const* d_in, const int* in_sizes, int n_in,
                              void* d_out, int out_size, void* d_ws, size_t ws_size,
                              hipStream_t stream) {
  (void)in_sizes; (void)n_in; (void)out_size; (void)ws_size;
  const float* x     = (const float*)d_in[0];
  const float* pred0 = (const float*)d_in[1];
  const float* gate0 = (const float*)d_in[2];
  const float* W_ih  = (const float*)d_in[3];
  const float* W_hh  = (const float*)d_in[4];
  const float* b_ih  = (const float*)d_in[5];
  const float* b_hh  = (const float*)d_in[6];
  const float* W_o   = (const float*)d_in[7];
  const float* b_o   = (const float*)d_in[8];
  const float* W1    = (const float*)d_in[9];
  const float* b1    = (const float*)d_in[10];
  const float* W2    = (const float*)d_in[11];
  const float* b2    = (const float*)d_in[12];
  const float* Wg    = (const float*)d_in[13];
  const float* bg    = (const float*)d_in[14];
  const float* Wa_ih = (const float*)d_in[15];
  const float* Wa_hh = (const float*)d_in[16];
  const float* ba_ih = (const float*)d_in[17];
  const float* ba_hh = (const float*)d_in[18];

  // scratch layout
  unsigned short* w1r  = (unsigned short*)d_ws;                        // 327,680 B
  float* werr  = (float*)((char*)d_ws + 327680);                       //   5,120 B
  float* initc = (float*)((char*)d_ws + 332800);                       //   5,120 B
  unsigned short* wexp = (unsigned short*)((char*)d_ws + 337920);      //  32,768 B
  unsigned short* wagt = (unsigned short*)((char*)d_ws + 370688);      //   8,192 B
  unsigned short* w2r  = (unsigned short*)((char*)d_ws + 378880);      //   2,048 B

  prep_w1r <<<NFRAG * 512 / THREADS, THREADS, 0, stream>>>(W1, w1r);
  prep_wexp<<<32 * 512 / THREADS, THREADS, 0, stream>>>(W_hh, wexp);
  prep_wagt<<<8 * 512 / THREADS, THREADS, 0, stream>>>(Wa_ih, Wa_hh, wagt);
  prep_w2r <<<2 * 512 / THREADS, THREADS, 0, stream>>>(W2, w2r);
  prep_misc<<<1, THREADS, 0, stream>>>(W1, werr, initc);

  mmoe_kernel<<<B_TOT / TILE, THREADS, 0, stream>>>(
      x, pred0, gate0, W_ih, b_ih, b_hh, W_o, b_o,
      b1, b2, Wg, bg, ba_ih, ba_hh,
      w1r, wexp, wagt, w2r, werr, initc, (float*)d_out);
}
